// ARWKV_7_83459804496093
// MI455X (gfx1250) — matrix operand order repaired and checked by reading, not compiled, not run on hardware
//
#include <hip/hip_runtime.h>
#include <math.h>

// ---------------- problem constants ----------------
#define BB   2
#define TT   1024
#define CC   2048
#define HH   32
#define NNd  64
#define MTOK (BB*TT)          // 2048 tokens
#define DFN  8192
#define D_DECAY 64
#define D_A     64
#define D_G     128

typedef __bf16 bf16;
typedef __attribute__((ext_vector_type(8)))  bf16  v8bf;
typedef __attribute__((ext_vector_type(16))) bf16  v16bf;
typedef __attribute__((ext_vector_type(8)))  float v8f;
typedef __attribute__((ext_vector_type(4)))  int   v4i;

// -------- CDNA5 async global->LDS path (guarded; sync fallback) --------
#if __has_builtin(__builtin_amdgcn_global_load_async_to_lds_b128) && \
    __has_builtin(__builtin_amdgcn_s_wait_asynccnt)
#define ASYNC_LDS 1
#endif

__device__ __forceinline__ void async_cp16(const bf16* g, bf16* l) {
#ifdef ASYNC_LDS
    // builtin signature (from hipcc diagnostic): first arg is v4i* (128-bit vector ptr)
    __builtin_amdgcn_global_load_async_to_lds_b128((v4i*)g, (v4i*)l, 0, 0);
#else
    *(v8bf*)l = *(const v8bf*)g;
#endif
}

template<int N>
__device__ __forceinline__ void wait_async() {
#ifdef ASYNC_LDS
    __builtin_amdgcn_s_wait_asynccnt(N);
#endif
}

// ---------------- block reduction helper ----------------
__device__ __forceinline__ float block_reduce_sum(float v, float* sbuf) {
    int tid = threadIdx.x;
    sbuf[tid] = v;
    __syncthreads();
    for (int s = 128; s > 0; s >>= 1) {
        if (tid < s) sbuf[tid] += sbuf[tid + s];
        __syncthreads();
    }
    float r = sbuf[0];
    __syncthreads();
    return r;
}

// ---------------- f32 -> bf16 convert ----------------
__global__ void k_cvt_bf16(const float* __restrict__ in, bf16* __restrict__ out, long n) {
    long i = (long)blockIdx.x * blockDim.x + threadIdx.x;
    if (i < n) out[i] = (bf16)in[i];
}

// ---------------- prep: rmsnorm(ln1) + token shift -> 6 bf16 mixes ----------------
__global__ __launch_bounds__(256)
void k_prep(const float* __restrict__ x, const float* __restrict__ x_prev,
            const float* __restrict__ ln1,
            const float* __restrict__ mr, const float* __restrict__ mw,
            const float* __restrict__ mk, const float* __restrict__ mv,
            const float* __restrict__ ma, const float* __restrict__ mg,
            bf16* __restrict__ xr, bf16* __restrict__ xw, bf16* __restrict__ xk,
            bf16* __restrict__ xv, bf16* __restrict__ xa, bf16* __restrict__ xg)
{
    __shared__ float sbuf[256];
    int bt = blockIdx.x;
    int b  = bt / TT;
    int t  = bt % TT;
    int tid = threadIdx.x;
    const float* xc = x + (size_t)bt * CC;

    float xl[8];
    float s = 0.f;
    #pragma unroll
    for (int i = 0; i < 8; ++i) { xl[i] = xc[tid*8 + i]; s += xl[i]*xl[i]; }
    float vs = block_reduce_sum(s, sbuf);
    float rs = rsqrtf(vs / (float)CC + 1e-6f);

    float pl[8];
    if (t == 0) {
        #pragma unroll
        for (int i = 0; i < 8; ++i) pl[i] = x_prev[(size_t)b*CC + tid*8 + i];
    } else {
        const float* xp = x + (size_t)(bt - 1) * CC;
        float sp = 0.f;
        #pragma unroll
        for (int i = 0; i < 8; ++i) { pl[i] = xp[tid*8 + i]; sp += pl[i]*pl[i]; }
        float vp = block_reduce_sum(sp, sbuf);
        float rp = rsqrtf(vp / (float)CC + 1e-6f);
        #pragma unroll
        for (int i = 0; i < 8; ++i) pl[i] = pl[i] * rp * ln1[tid*8 + i];
    }

    size_t base = (size_t)bt * CC + tid*8;
    #pragma unroll
    for (int i = 0; i < 8; ++i) {
        int c = tid*8 + i;
        float xn = xl[i] * rs * ln1[c];
        float sx = pl[i] - xn;
        xr[base+i] = (bf16)(xn + sx*mr[c]);
        xw[base+i] = (bf16)(xn + sx*mw[c]);
        xk[base+i] = (bf16)(xn + sx*mk[c]);
        xv[base+i] = (bf16)(xn + sx*mv[c]);
        xa[base+i] = (bf16)(xn + sx*ma[c]);
        xg[base+i] = (bf16)(xn + sx*mg[c]);
    }
}

// ---------------- generic bf16 WMMA GEMM (async double-buffered) ----------------
// C[M,N] (f32) = A[M,K] (bf16 row-major) x B
//   BKN == 0 : B is [N,K] (nn.Linear weight [out,in], out = A @ W^T) -- async B
//   BKN == 1 : B is [K,N] (out = A @ W)                              -- sync transposed B
// Block tile 128x128, 8 waves, wave tile 64x32 (4x2 wmma 16x16x32 tiles).
// Double-buffered LDS; next K-tile loads issued async (ASYNCcnt) and overlapped
// with the current tile's 8 WMMAs; s_wait_asynccnt <= inflight isolates them.
template<int BKN>
__global__ __launch_bounds__(256)
void k_gemm_bf16(const bf16* __restrict__ A, const bf16* __restrict__ Bw,
                 float* __restrict__ Cout, int M, int Nn, int K)
{
    __shared__ bf16 As[2][128][40];
    __shared__ bf16 Bs[2][128][40];

    const int tid  = threadIdx.x;
    const int lane = tid & 31;
    const int wid  = tid >> 5;
    const int wm   = wid & 1;       // 2 waves along M
    const int wn   = wid >> 1;      // 4 waves along N
    const int blockN = blockIdx.x * 128;
    const int blockM = blockIdx.y * 128;
    const int nk = K >> 5;          // K always a multiple of 32 here

    // cooperative-load coordinates: 2 x 16B per thread per 128x32 tile
    int aM[2], aK[2];
    #pragma unroll
    for (int i = 0; i < 2; ++i) {
        int lin = (tid*2 + i) * 8;
        aM[i] = lin >> 5;
        aK[i] = lin & 31;
    }

    v8f acc[4][2];
    #pragma unroll
    for (int mi = 0; mi < 4; ++mi)
        #pragma unroll
        for (int ni = 0; ni < 2; ++ni) { v8f z = {}; acc[mi][ni] = z; }

    const int mrow = lane & 15;
    const int kA   = (lane < 16) ? 0 : 8;    // A: lanes0-15 K 0..7/16..23, lanes16-31 K 8..15/24..31
    const int kB   = (lane < 16) ? 0 : 8;    // B: lanes0-15 K 0..7/16..23, lanes16-31 K 8..15/24..31

    // tile loader: OOB rows/cols are clamped (not zeroed) so every async op is
    // issued with a uniform EXEC mask and the in-flight ASYNCcnt stays constant;
    // the garbage only feeds accumulator slots the guarded epilogue never stores.
    auto issue_tile = [&](int kt, int bufi) {
        const int k0 = kt * 32;
        #pragma unroll
        for (int i = 0; i < 2; ++i) {
            int gm = blockM + aM[i]; if (gm >= M) gm = M - 1;
            async_cp16(A + (size_t)gm * K + k0 + aK[i], &As[bufi][aM[i]][aK[i]]);
        }
        if (BKN == 0) {
            #pragma unroll
            for (int i = 0; i < 2; ++i) {
                int gn = blockN + aM[i]; if (gn >= Nn) gn = Nn - 1;
                async_cp16(Bw + (size_t)gn * K + k0 + aK[i], &Bs[bufi][aM[i]][aK[i]]);
            }
        } else {
            // B is [K,N]: contiguous along n -> transpose-scatter via LDS stores (DScnt)
            #pragma unroll
            for (int i = 0; i < 2; ++i) {
                int lin = tid*2 + i;             // 32 k-rows x 16 n-octets
                int kk = lin >> 4;
                int nv = (lin & 15) * 8;
                int gn = blockN + nv;
                v8bf val = {};
                if (gn + 8 <= Nn) val = *(const v8bf*)(Bw + (size_t)(k0 + kk) * Nn + gn);
                #pragma unroll
                for (int j = 0; j < 8; ++j) Bs[bufi][nv + j][kk] = val[j];
            }
        }
    };

    int bufi = 0;
    issue_tile(0, 0);

    for (int kt = 0; kt < nk; ++kt) {
        const int nxt = bufi ^ 1;
        if (kt + 1 < nk) {
            issue_tile(kt + 1, nxt);            // overlaps with this tile's WMMAs
            wait_async<(BKN == 0) ? 4 : 2>();   // wait current tile; next stays in flight
        } else {
            wait_async<0>();
        }
        __syncthreads();                        // also waits DScnt for sync B path

        // --- build fragments from LDS ---
        v16bf afr[4];
        #pragma unroll
        for (int mi = 0; mi < 4; ++mi) {
            int row = wm*64 + mi*16 + mrow;
            v8bf lo = *(const v8bf*)(&As[bufi][row][kA]);
            v8bf hi = *(const v8bf*)(&As[bufi][row][kA + 16]);
            afr[mi] = __builtin_shufflevector(lo, hi, 0,1,2,3,4,5,6,7,8,9,10,11,12,13,14,15);
        }
        v16bf bfr[2];
        #pragma unroll
        for (int ni = 0; ni < 2; ++ni) {
            int col = wn*32 + ni*16 + mrow;
            v8bf lo = *(const v8bf*)(&Bs[bufi][col][kB]);
            v8bf hi = *(const v8bf*)(&Bs[bufi][col][kB + 16]);
            bfr[ni] = __builtin_shufflevector(lo, hi, 0,1,2,3,4,5,6,7,8,9,10,11,12,13,14,15);
        }

        // --- matrix multiply-accumulate ---
        #pragma unroll
        for (int mi = 0; mi < 4; ++mi)
            #pragma unroll
            for (int ni = 0; ni < 2; ++ni)
                acc[mi][ni] = __builtin_amdgcn_wmma_f32_16x16x32_bf16(
                    false, afr[mi], false, bfr[ni], (short)0, acc[mi][ni], false, false);
        __syncthreads();
        bufi = nxt;
    }

    // --- store C ---
    const int rowoff = (lane < 16) ? 0 : 8;
    #pragma unroll
    for (int mi = 0; mi < 4; ++mi) {
        #pragma unroll
        for (int ni = 0; ni < 2; ++ni) {
            int m0 = blockM + wm*64 + mi*16;
            int n0 = blockN + wn*32 + ni*16;
            int c  = n0 + (lane & 15);
            if (c < Nn) {
                #pragma unroll
                for (int i = 0; i < 8; ++i) {
                    int r = m0 + i + rowoff;
                    if (r < M) Cout[(size_t)r * Nn + c] = acc[mi][ni][i];
                }
            }
        }
    }
}

// ---------------- element-wise kernels ----------------
__global__ void k_tanh_bf16(const float* __restrict__ in, bf16* __restrict__ out, long n) {
    long i = (long)blockIdx.x * blockDim.x + threadIdx.x;
    if (i < n) out[i] = (bf16)tanhf(in[i]);
}
__global__ void k_sig_bf16(const float* __restrict__ in, bf16* __restrict__ out, long n) {
    long i = (long)blockIdx.x * blockDim.x + threadIdx.x;
    if (i < n) out[i] = (bf16)(1.0f / (1.0f + expf(-in[i])));
}
// decay = exp(-exp(-softplus(-(w0 + hw)) - 0.5))
__global__ void k_wdecay(const float* __restrict__ hw, const float* __restrict__ w0,
                         float* __restrict__ decay, long n) {
    long i = (long)blockIdx.x * blockDim.x + threadIdx.x;
    if (i < n) {
        int c = (int)(i & (CC - 1));
        float u    = w0[c] + hw[i];
        float wpre = -log1pf(expf(-u)) - 0.5f;
        decay[i]   = expf(-expf(wpre));
    }
}
__global__ void k_asig(const float* __restrict__ ha2, const float* __restrict__ a0,
                       float* __restrict__ a, long n) {
    long i = (long)blockIdx.x * blockDim.x + threadIdx.x;
    if (i < n) {
        int c = (int)(i & (CC - 1));
        a[i] = 1.0f / (1.0f + expf(-(a0[c] + ha2[i])));
    }
}

// per-head kk normalization + k modulation -> kmod, aa, bb
__global__ __launch_bounds__(256)
void k_post1(const float* __restrict__ kin, const float* __restrict__ a,
             const float* __restrict__ kkw, const float* __restrict__ kaw,
             float* __restrict__ kmod, float* __restrict__ aa, float* __restrict__ bb)
{
    __shared__ float part[256];
    __shared__ float hinv[HH];
    size_t base = (size_t)blockIdx.x * CC;
    int tid = threadIdx.x;
    float kv[8], av[8], kkv[8];
    float s = 0.f;
    #pragma unroll
    for (int i = 0; i < 8; ++i) {
        int c = tid*8 + i;
        kv[i]  = kin[base + c];
        av[i]  = a[base + c];
        kkv[i] = kv[i] * kkw[c];
        s += kkv[i] * kkv[i];
    }
    part[tid] = s;
    __syncthreads();
    if (tid < HH) {
        float t = 0.f;
        #pragma unroll
        for (int j = 0; j < 8; ++j) t += part[tid*8 + j];
        hinv[tid] = 1.0f / fmaxf(sqrtf(t), 1e-12f);
    }
    __syncthreads();
    #pragma unroll
    for (int i = 0; i < 8; ++i) {
        int c = tid*8 + i;
        int h = c >> 6;
        float kkn = kkv[i] * hinv[h];
        kmod[base + c] = kv[i] * (1.0f + (av[i] - 1.0f) * kaw[c]);
        aa[base + c]   = -kkn;
        bb[base + c]   = kkn * av[i];
    }
}

// ---------------- RWKV-7 recurrent scan ----------------
// one block per (b,h); 64 threads; thread tid owns state row v=tid (64 f32 in regs)
__global__ __launch_bounds__(64)
void k_scan(const float* __restrict__ r, const float* __restrict__ d,
            const float* __restrict__ k, const float* __restrict__ v,
            const float* __restrict__ aa, const float* __restrict__ bb,
            const float* __restrict__ state, float* __restrict__ xx)
{
    int bh = blockIdx.x;
    int b  = bh / HH;
    int h  = bh % HH;
    int tid = threadIdx.x;

    float S[NNd];
    const float* st = state + ((size_t)(b*HH + h) * NNd + tid) * NNd;
    #pragma unroll
    for (int c = 0; c < NNd; ++c) S[c] = st[c];

    __shared__ float shr[NNd], shd[NNd], shk[NNd], shv[NNd], sha[NNd], shb[NNd];

    for (int t = 0; t < TT; ++t) {
        size_t base = ((size_t)(b*TT + t) * HH + h) * NNd;
        shr[tid] = r[base + tid];
        shd[tid] = d[base + tid];
        shk[tid] = k[base + tid];
        shv[tid] = v[base + tid];
        sha[tid] = aa[base + tid];
        shb[tid] = bb[base + tid];
        __syncthreads();

        float sa = 0.f;
        #pragma unroll
        for (int c = 0; c < NNd; ++c) sa += S[c] * sha[c];
        float vv = shv[tid];
        float o  = 0.f;
        #pragma unroll
        for (int c = 0; c < NNd; ++c) {
            float sc = S[c] * shd[c] + sa * shb[c] + vv * shk[c];
            S[c] = sc;
            o += sc * shr[c];
        }
        xx[base + tid] = o;
        __syncthreads();
    }
}

// bonus + gate -> bf16 attention-projection input
__global__ __launch_bounds__(256)
void k_post2(const float* __restrict__ r, const float* __restrict__ kmod,
             const float* __restrict__ v, const float* __restrict__ xx,
             const float* __restrict__ g, const float* __restrict__ rkw,
             bf16* __restrict__ attin)
{
    __shared__ float part[256];
    __shared__ float hs[HH];
    size_t base = (size_t)blockIdx.x * CC;
    int tid = threadIdx.x;
    float s = 0.f;
    #pragma unroll
    for (int i = 0; i < 8; ++i) {
        int c = tid*8 + i;
        s += r[base + c] * kmod[base + c] * rkw[c];
    }
    part[tid] = s;
    __syncthreads();
    if (tid < HH) {
        float t = 0.f;
        #pragma unroll
        for (int j = 0; j < 8; ++j) t += part[tid*8 + j];
        hs[tid] = t;
    }
    __syncthreads();
    #pragma unroll
    for (int i = 0; i < 8; ++i) {
        int c = tid*8 + i;
        int h = c >> 6;
        float bonus = hs[h] * v[base + c];
        attin[base + c] = (bf16)((xx[base + c] + bonus) * g[base + c]);
    }
}

// residual add + rmsnorm(ln2) -> xatt f32, xn2 bf16
__global__ __launch_bounds__(256)
void k_resid_ln2(const float* __restrict__ x, const float* __restrict__ att,
                 const float* __restrict__ ln2,
                 float* __restrict__ xatt, bf16* __restrict__ xn2)
{
    __shared__ float sbuf[256];
    size_t base = (size_t)blockIdx.x * CC;
    int tid = threadIdx.x;
    float xl[8];
    float s = 0.f;
    #pragma unroll
    for (int i = 0; i < 8; ++i) {
        int c = tid*8 + i;
        xl[i] = x[base + c] + att[base + c];
        s += xl[i] * xl[i];
    }
    float vs = block_reduce_sum(s, sbuf);
    float rs = rsqrtf(vs / (float)CC + 1e-6f);
    #pragma unroll
    for (int i = 0; i < 8; ++i) {
        int c = tid*8 + i;
        xatt[base + c] = xl[i];
        xn2[base + c]  = (bf16)(xl[i] * rs * ln2[c]);
    }
}

__global__ void k_swiglu(const float* __restrict__ hg, const float* __restrict__ hu,
                         bf16* __restrict__ out, long n) {
    long i = (long)blockIdx.x * blockDim.x + threadIdx.x;
    if (i < n) {
        float z = hg[i];
        out[i] = (bf16)((z / (1.0f + expf(-z))) * hu[i]);
    }
}
__global__ void k_final(const float* __restrict__ xatt, const float* __restrict__ mlp,
                        float* __restrict__ out, long n) {
    long i = (long)blockIdx.x * blockDim.x + threadIdx.x;
    if (i < n) out[i] = xatt[i] + mlp[i];
}

// ---------------- host-side orchestration ----------------
static inline dim3 gemm_grid(int M, int N) { return dim3((N + 127) / 128, (M + 127) / 128); }

extern "C" void kernel_launch(void* const* d_in, const int* in_sizes, int n_in,
                              void* d_out, int out_size, void* d_ws, size_t ws_size,
                              hipStream_t stream) {
    (void)in_sizes; (void)n_in; (void)out_size; (void)ws_size;

    const float* x      = (const float*)d_in[0];
    const float* x_prev = (const float*)d_in[1];
    const float* state  = (const float*)d_in[2];
    const float* ln1_w  = (const float*)d_in[3];
    const float* ln2_w  = (const float*)d_in[4];
    const float* mx_r   = (const float*)d_in[5];
    const float* mx_w   = (const float*)d_in[6];
    const float* mx_k   = (const float*)d_in[7];
    const float* mx_v   = (const float*)d_in[8];
    const float* mx_a   = (const float*)d_in[9];
    const float* mx_g   = (const float*)d_in[10];
    const float* w0     = (const float*)d_in[11];
    const float* w1     = (const float*)d_in[12];
    const float* w2     = (const float*)d_in[13];
    const float* a0     = (const float*)d_in[14];
    const float* a1     = (const float*)d_in[15];
    const float* a2     = (const float*)d_in[16];
    const float* g1     = (const float*)d_in[17];
    const float* g2     = (const float*)d_in[18];
    const float* k_k    = (const float*)d_in[19];
    const float* k_a    = (const float*)d_in[20];
    const float* r_k    = (const float*)d_in[21];
    const float* R_     = (const float*)d_in[22];
    const float* K_     = (const float*)d_in[23];
    const float* V_     = (const float*)d_in[24];
    const float* O_     = (const float*)d_in[25];
    const float* gate_w = (const float*)d_in[26];
    const float* up_w   = (const float*)d_in[27];
    const float* down_w = (const float*)d_in[28];
    float* out = (float*)d_out;

    // workspace carve-up
    char* p = (char*)d_ws;
    auto alloc = [&](size_t bytes) -> void* {
        void* q = (void*)p;
        p += (bytes + 255) & ~(size_t)255;
        return q;
    };
    const size_t MC  = (size_t)MTOK * CC;
    const size_t MF  = (size_t)MTOK * DFN;

    bf16* Rb    = (bf16*)alloc((size_t)CC*CC*2);
    bf16* Kb    = (bf16*)alloc((size_t)CC*CC*2);
    bf16* Vb    = (bf16*)alloc((size_t)CC*CC*2);
    bf16* Ob    = (bf16*)alloc((size_t)CC*CC*2);
    bf16* w1b   = (bf16*)alloc((size_t)CC*D_DECAY*2);
    bf16* w2b   = (bf16*)alloc((size_t)D_DECAY*CC*2);
    bf16* a1b   = (bf16*)alloc((size_t)CC*D_A*2);
    bf16* a2b   = (bf16*)alloc((size_t)D_A*CC*2);
    bf16* g1b   = (bf16*)alloc((size_t)CC*D_G*2);
    bf16* g2b   = (bf16*)alloc((size_t)D_G*CC*2);
    bf16* gateb = (bf16*)alloc((size_t)DFN*CC*2);
    bf16* upb   = (bf16*)alloc((size_t)DFN*CC*2);
    bf16* downb = (bf16*)alloc((size_t)CC*DFN*2);

    bf16* xrB = (bf16*)alloc(MC*2);
    bf16* xwB = (bf16*)alloc(MC*2);
    bf16* xkB = (bf16*)alloc(MC*2);
    bf16* xvB = (bf16*)alloc(MC*2);
    bf16* xaB = (bf16*)alloc(MC*2);
    bf16* xgB = (bf16*)alloc(MC*2);

    float* rF    = (float*)alloc(MC*4);
    float* kF    = (float*)alloc(MC*4);
    float* vF    = (float*)alloc(MC*4);
    float* hw1   = (float*)alloc((size_t)MTOK*D_DECAY*4);
    bf16*  hw1b  = (bf16*)alloc((size_t)MTOK*D_DECAY*2);
    float* hwF   = (float*)alloc(MC*4);
    float* decay = (float*)alloc(MC*4);
    float* ha1   = (float*)alloc((size_t)MTOK*D_A*4);
    bf16*  ha1b  = (bf16*)alloc((size_t)MTOK*D_A*2);
    float* ha2   = (float*)alloc(MC*4);
    float* aF    = (float*)alloc(MC*4);
    float* hg1   = (float*)alloc((size_t)MTOK*D_G*4);
    bf16*  hg1b  = (bf16*)alloc((size_t)MTOK*D_G*2);
    float* gF    = (float*)alloc(MC*4);
    float* kmod  = (float*)alloc(MC*4);
    float* aaF   = (float*)alloc(MC*4);
    float* bbF   = (float*)alloc(MC*4);
    float* xxF   = (float*)alloc(MC*4);
    bf16*  attinB= (bf16*)alloc(MC*2);
    float* attF  = (float*)alloc(MC*4);
    float* xattF = (float*)alloc(MC*4);
    bf16*  xn2B  = (bf16*)alloc(MC*2);
    float* hgate = (float*)alloc(MF*4);
    float* hup   = (float*)alloc(MF*4);
    bf16*  hswiB = (bf16*)alloc(MF*2);
    float* mlpF  = (float*)alloc(MC*4);

    auto cvt = [&](const float* src, bf16* dst, long n) {
        k_cvt_bf16<<<dim3((unsigned)((n + 255) / 256)), dim3(256), 0, stream>>>(src, dst, n);
    };

    // 1) weight conversion f32 -> bf16
    cvt(R_, Rb, (long)CC*CC);   cvt(K_, Kb, (long)CC*CC);
    cvt(V_, Vb, (long)CC*CC);   cvt(O_, Ob, (long)CC*CC);
    cvt(w1, w1b, (long)CC*D_DECAY);  cvt(w2, w2b, (long)D_DECAY*CC);
    cvt(a1, a1b, (long)CC*D_A);      cvt(a2, a2b, (long)D_A*CC);
    cvt(g1, g1b, (long)CC*D_G);      cvt(g2, g2b, (long)D_G*CC);
    cvt(gate_w, gateb, (long)DFN*CC);
    cvt(up_w,   upb,   (long)DFN*CC);
    cvt(down_w, downb, (long)CC*DFN);

    // 2) prep: ln1 + token shift
    k_prep<<<dim3(MTOK), dim3(256), 0, stream>>>(x, x_prev, ln1_w,
        mx_r, mx_w, mx_k, mx_v, mx_a, mx_g, xrB, xwB, xkB, xvB, xaB, xgB);

    // 3) big projections r/k/v  (W is [out,in] -> BKN=0)
    k_gemm_bf16<0><<<gemm_grid(MTOK, CC), dim3(256), 0, stream>>>(xrB, Rb, rF, MTOK, CC, CC);
    k_gemm_bf16<0><<<gemm_grid(MTOK, CC), dim3(256), 0, stream>>>(xkB, Kb, kF, MTOK, CC, CC);
    k_gemm_bf16<0><<<gemm_grid(MTOK, CC), dim3(256), 0, stream>>>(xvB, Vb, vF, MTOK, CC, CC);

    // 4) decay LoRA: tanh(xw@w1)@w2 ; w1 is [C,D] -> BKN=1
    k_gemm_bf16<1><<<gemm_grid(MTOK, D_DECAY), dim3(256), 0, stream>>>(xwB, w1b, hw1, MTOK, D_DECAY, CC);
    k_tanh_bf16<<<dim3((unsigned)(((long)MTOK*D_DECAY + 255)/256)), dim3(256), 0, stream>>>(hw1, hw1b, (long)MTOK*D_DECAY);
    k_gemm_bf16<1><<<gemm_grid(MTOK, CC), dim3(256), 0, stream>>>(hw1b, w2b, hwF, MTOK, CC, D_DECAY);
    k_wdecay<<<dim3((unsigned)((MC + 255)/256)), dim3(256), 0, stream>>>(hwF, w0, decay, (long)MC);

    // 5) a LoRA: sigmoid(a0 + (xa@a1)@a2)
    k_gemm_bf16<1><<<gemm_grid(MTOK, D_A), dim3(256), 0, stream>>>(xaB, a1b, ha1, MTOK, D_A, CC);
    cvt(ha1, ha1b, (long)MTOK*D_A);
    k_gemm_bf16<1><<<gemm_grid(MTOK, CC), dim3(256), 0, stream>>>(ha1b, a2b, ha2, MTOK, CC, D_A);
    k_asig<<<dim3((unsigned)((MC + 255)/256)), dim3(256), 0, stream>>>(ha2, a0, aF, (long)MC);

    // 6) g LoRA: sigmoid(xg@g1)@g2
    k_gemm_bf16<1><<<gemm_grid(MTOK, D_G), dim3(256), 0, stream>>>(xgB, g1b, hg1, MTOK, D_G, CC);
    k_sig_bf16<<<dim3((unsigned)(((long)MTOK*D_G + 255)/256)), dim3(256), 0, stream>>>(hg1, hg1b, (long)MTOK*D_G);
    k_gemm_bf16<1><<<gemm_grid(MTOK, CC), dim3(256), 0, stream>>>(hg1b, g2b, gF, MTOK, CC, D_G);

    // 7) kk head-norm, k modulation, aa/bb
    k_post1<<<dim3(MTOK), dim3(256), 0, stream>>>(kF, aF, k_k, k_a, kmod, aaF, bbF);

    // 8) sequential RWKV-7 scan: one block per (b,h)
    k_scan<<<dim3(BB*HH), dim3(64), 0, stream>>>(rF, decay, kmod, vF, aaF, bbF, state, xxF);

    // 9) bonus + gate -> bf16 input for output projection
    k_post2<<<dim3(MTOK), dim3(256), 0, stream>>>(rF, kmod, vF, xxF, gF, r_k, attinB);

    // 10) output projection (O_ is [out,in] -> BKN=0)
    k_gemm_bf16<0><<<gemm_grid(MTOK, CC), dim3(256), 0, stream>>>(attinB, Ob, attF, MTOK, CC, CC);

    // 11) residual + ln2
    k_resid_ln2<<<dim3(MTOK), dim3(256), 0, stream>>>(x, attF, ln2_w, xattF, xn2B);

    // 12) SwiGLU MLP
    k_gemm_bf16<0><<<gemm_grid(MTOK, DFN), dim3(256), 0, stream>>>(xn2B, gateb, hgate, MTOK, DFN, CC);
    k_gemm_bf16<0><<<gemm_grid(MTOK, DFN), dim3(256), 0, stream>>>(xn2B, upb,   hup,   MTOK, DFN, CC);
    k_swiglu<<<dim3((unsigned)((MF + 255)/256)), dim3(256), 0, stream>>>(hgate, hup, hswiB, (long)MF);
    k_gemm_bf16<0><<<gemm_grid(MTOK, CC), dim3(256), 0, stream>>>(hswiB, downb, mlpF, MTOK, CC, DFN);

    // 13) final residual
    k_final<<<dim3((unsigned)((MC + 255)/256)), dim3(256), 0, stream>>>(xattF, mlpF, out, (long)MC);
}
